// CrossAttentionBridge_76553497084523
// MI455X (gfx1250) — matrix operand order repaired and checked by reading, not compiled, not run on hardware
//
#include <hip/hip_runtime.h>

// ---------------------------------------------------------------------------
// CrossAttentionBridge on MI455X (gfx1250), wave32, bf16 WMMA w/ f32 accum.
// B=2, L=2048, d=512, H=8, hd=64.  ~22.5 GFLOP, HBM floor ~24MB -> compute on
// v_wmma_f32_16x16x32_bf16; flash-style attention (no 2048x2048 score tensor).
// ---------------------------------------------------------------------------

typedef __attribute__((ext_vector_type(16))) __bf16 v16bf;
typedef __attribute__((ext_vector_type(8)))  float  v8f;

#define D_MODEL 512
#define SEQ     2048
#define ROWS    4096      // B * L
#define NHEAD   8
#define HDIM    64

static __device__ __forceinline__ v8f wmma_bf16(v16bf a, v16bf b, v8f c) {
  // D = A(16x32) * B(32x16) + C(16x16 f32)
  return __builtin_amdgcn_wmma_f32_16x16x32_bf16(false, a, false, b,
                                                 (short)0, c, false, false);
}

// A-fragment (16x32, M=lane%16): lane holds K chunk [k0+16*hl .. +15]
// (hl = lane>>4).  Two 16B loads.
static __device__ __forceinline__ v16bf load_afrag(const __bf16* row, int k0, int hl) {
  union { uint4 q[2]; v16bf v; } u;
  const uint4* p = (const uint4*)(row + k0 + 16 * hl);
  u.q[0] = p[0];
  u.q[1] = p[1];
  return u.v;
}

// B-fragment (32x16, N=lane%16): lane holds K chunk [k0+16*hl .. +15].
static __device__ __forceinline__ v16bf load_bfrag(const __bf16* row, int k0, int hl) {
  union { uint4 q[2]; v16bf v; } u;
  const uint4* p = (const uint4*)(row + k0 + 16 * hl);
  u.q[0] = p[0];
  u.q[1] = p[1];
  return u.v;
}

// 16x16 output tile of C = A * W^T over K=512; At/Wt already offset to this
// lane's row (tile_row + lane%16).
static __device__ __forceinline__ v8f gemm_tile_512(const __bf16* At, const __bf16* Wt, int hl) {
  v8f acc = {0.f, 0.f, 0.f, 0.f, 0.f, 0.f, 0.f, 0.f};
#pragma unroll
  for (int k = 0; k < 512; k += 32)
    acc = wmma_bf16(load_afrag(At, k, hl), load_bfrag(Wt, k, hl), acc);
  return acc;
}

// --------------------------- small prep kernels ----------------------------

__global__ void __launch_bounds__(256)
k_cvt(const float* __restrict__ s, __bf16* __restrict__ d, int n) {
  int i = blockIdx.x * 256 + threadIdx.x;
  if (i < n) d[i] = (__bf16)s[i];
}

// bias[i,j]: bilinear upsample (half-pixel, edge-clamped) of
// g(|a-b|) = exp(-0.1 d) - 0.05 d on a 128x128 grid, to 2048x2048.
__global__ void __launch_bounds__(256)
k_bias(float* __restrict__ BIAS) {
  int idx = blockIdx.x * 256 + threadIdx.x;
  int i = idx >> 11, j = idx & 2047;
  float x = fmaxf((i + 0.5f) * (1.0f / 16.0f) - 0.5f, 0.f);
  float y = fmaxf((j + 0.5f) * (1.0f / 16.0f) - 0.5f, 0.f);
  int x0 = (int)x; float fx = x - (float)x0; int x1 = min(x0 + 1, 127);
  int y0 = (int)y; float fy = y - (float)y0; int y1 = min(y0 + 1, 127);
  float d00 = fabsf((float)(x0 - y0)), d01 = fabsf((float)(x0 - y1));
  float d10 = fabsf((float)(x1 - y0)), d11 = fabsf((float)(x1 - y1));
  float g00 = expf(-0.1f * d00) - 0.05f * d00;
  float g01 = expf(-0.1f * d01) - 0.05f * d01;
  float g10 = expf(-0.1f * d10) - 0.05f * d10;
  float g11 = expf(-0.1f * d11) - 0.05f * d11;
  BIAS[idx] = (1.f - fx) * ((1.f - fy) * g00 + fy * g01) +
              fx * ((1.f - fy) * g10 + fy * g11);
}

// Fused layernorm of decoder (rows 0..4095) and encoder (rows 4096..8191),
// one wave per 512-wide row, bf16 output.
__global__ void __launch_bounds__(256)
k_ln(const float* __restrict__ dec, const float* __restrict__ enc,
     const float* __restrict__ g, const float* __restrict__ bt,
     __bf16* __restrict__ LNQ, __bf16* __restrict__ LNKV) {
  int row  = blockIdx.x * 8 + (threadIdx.x >> 5);
  int lane = threadIdx.x & 31;
  const float* src; __bf16* dst;
  if (row < ROWS) { src = dec + (size_t)row * D_MODEL;          dst = LNQ  + (size_t)row * D_MODEL; }
  else            { src = enc + (size_t)(row - ROWS) * D_MODEL; dst = LNKV + (size_t)(row - ROWS) * D_MODEL; }
  float4 x[4]; float s = 0.f, s2 = 0.f;
#pragma unroll
  for (int i = 0; i < 4; i++) {
    x[i] = *(const float4*)(src + i * 128 + lane * 4);
    s  += x[i].x + x[i].y + x[i].z + x[i].w;
    s2 += x[i].x * x[i].x + x[i].y * x[i].y + x[i].z * x[i].z + x[i].w * x[i].w;
  }
#pragma unroll
  for (int m = 16; m >= 1; m >>= 1) { s += __shfl_xor(s, m, 32); s2 += __shfl_xor(s2, m, 32); }
  float mu  = s * (1.f / 512.f);
  float var = s2 * (1.f / 512.f) - mu * mu;
  float rs  = rsqrtf(var + 1e-5f);
#pragma unroll
  for (int i = 0; i < 4; i++) {
    int c = i * 128 + lane * 4;
    float xv[4] = {x[i].x, x[i].y, x[i].z, x[i].w};
    union { uint2 u; __bf16 h[4]; } pk;
#pragma unroll
    for (int jj = 0; jj < 4; jj++)
      pk.h[jj] = (__bf16)((xv[jj] - mu) * rs * g[c + jj] + bt[c + jj]);
    *(uint2*)(dst + c) = pk.u;
  }
}

// ------------------------------ QKV GEMM -----------------------------------
// which = gw>>13: 0 -> Q (pre-scaled by hd^-0.5), 1 -> K, 2 -> V (stored
// transposed per (b,h): VT[(bh*64+dh)*2048 + t]).
__global__ void __launch_bounds__(256)
k_qkv(const __bf16* __restrict__ LNQ, const __bf16* __restrict__ LNKV,
      const __bf16* __restrict__ WQKV,
      __bf16* __restrict__ QB, __bf16* __restrict__ KB, __bf16* __restrict__ VT) {
  int gw   = (blockIdx.x * 256 + threadIdx.x) >> 5;
  int lane = threadIdx.x & 31, qc = lane & 15, hl = lane >> 4;
  int which = gw >> 13;
  int rem   = gw & 8191;
  int mt = rem >> 5, nt = rem & 31;
  const __bf16* A  = (which == 0) ? LNQ : LNKV;
  const __bf16* At = A + (size_t)(mt * 16 + qc) * D_MODEL;
  const __bf16* Wt = WQKV + (size_t)(which * 512 + nt * 16 + qc) * D_MODEL;
  v8f acc = gemm_tile_512(At, Wt, hl);
#pragma unroll
  for (int r = 0; r < 8; r++) {
    int row = mt * 16 + r + 8 * hl;       // token index in [0, 4096)
    int col = nt * 16 + qc;               // feature in [0, 512)
    float v = acc[r];
    if (which == 0) {
      QB[(size_t)row * D_MODEL + col] = (__bf16)(v * 0.125f);   // 1/sqrt(64)
    } else if (which == 1) {
      KB[(size_t)row * D_MODEL + col] = (__bf16)v;
    } else {
      int h = col >> 6, dh = col & 63, bb = row >> 11, t = row & 2047;
      VT[(size_t)((bb * NHEAD + h) * HDIM + dh) * SEQ + t] = (__bf16)v;
    }
  }
}

// --------------------------- flash attention -------------------------------
// One wave per (b, h, 16-query tile).  Computes S^T = K * Q^T tiles (so the
// softmax column q == lane&15 -> per-lane stats + one shfl_xor(16)), then
// O^T += V^T * P^T.  8 WMMAs per 32-kv step, 512 per wave.
__global__ void __launch_bounds__(256)
k_attn(const __bf16* __restrict__ Q, const __bf16* __restrict__ K,
       const __bf16* __restrict__ VT, const float* __restrict__ BIAS,
       __bf16* __restrict__ ATT) {
  const int lane = threadIdx.x & 31;
  const int qc = lane & 15, hl = lane >> 4;
  const int gw = (blockIdx.x * 256 + threadIdx.x) >> 5;
  const int qt = gw & 127;
  const int bh = gw >> 7;                 // 0..15
  const int b  = bh >> 3, h = bh & 7;
  const bool lo = (lane < 16);

  const __bf16* qrow = Q + (size_t)(b * SEQ + qt * 16 + qc) * D_MODEL + h * HDIM;
  const v16bf qb0 = load_bfrag(qrow, 0, hl);    // Q^T B-frags, d = 0..31 / 32..63
  const v16bf qb1 = load_bfrag(qrow, 32, hl);

  const __bf16* Kbase  = K  + (size_t)b * SEQ * D_MODEL + h * HDIM;
  const __bf16* VTbase = VT + (size_t)bh * HDIM * SEQ;
  const float*  brow   = BIAS + (size_t)(qt * 16 + qc) * SEQ;

  const v8f vz = {0.f, 0.f, 0.f, 0.f, 0.f, 0.f, 0.f, 0.f};
  v8f o0 = vz, o1 = vz, o2 = vz, o3 = vz;   // O^T tiles, dh = 0..15/16..31/32..47/48..63
  float m_run = -1e30f, l_run = 0.f;

#pragma unroll 1
  for (int kt = 0; kt < SEQ; kt += 32) {
    const __bf16* kr0 = Kbase + (size_t)(kt + qc) * D_MODEL;
    const __bf16* kr1 = kr0 + 16 * D_MODEL;
    v8f s0 = vz, s1 = vz;                 // S^T subtiles: kv kt..+15, kt+16..+31
    s0 = wmma_bf16(load_afrag(kr0, 0,  hl), qb0, s0);
    s0 = wmma_bf16(load_afrag(kr0, 32, hl), qb1, s0);
    s1 = wmma_bf16(load_afrag(kr1, 0,  hl), qb0, s1);
    s1 = wmma_bf16(load_afrag(kr1, 32, hl), qb1, s1);

    // temporal bias: this lane's rows are kv = kt(+16) + 8*hl + r, col q fixed
    const float4* bp = (const float4*)(brow + kt + 8 * hl);
    float4 ba = bp[0], bb = bp[1], bc = bp[4], bd = bp[5];
    s0[0] += ba.x; s0[1] += ba.y; s0[2] += ba.z; s0[3] += ba.w;
    s0[4] += bb.x; s0[5] += bb.y; s0[6] += bb.z; s0[7] += bb.w;
    s1[0] += bc.x; s1[1] += bc.y; s1[2] += bc.z; s1[3] += bc.w;
    s1[4] += bd.x; s1[5] += bd.y; s1[6] += bd.z; s1[7] += bd.w;

    // online softmax over kv for column q (lane pair l, l^16)
    float tmax = s0[0];
#pragma unroll
    for (int r = 1; r < 8; r++) tmax = fmaxf(tmax, s0[r]);
#pragma unroll
    for (int r = 0; r < 8; r++) tmax = fmaxf(tmax, s1[r]);
    tmax = fmaxf(tmax, __shfl_xor(tmax, 16, 32));
    float m_new = fmaxf(m_run, tmax);
    float alpha = __expf(m_run - m_new);
    float p0[8], p1[8], s = 0.f;
#pragma unroll
    for (int r = 0; r < 8; r++) {
      p0[r] = __expf(s0[r] - m_new);
      p1[r] = __expf(s1[r] - m_new);
      s += p0[r] + p1[r];
    }
    s += __shfl_xor(s, 16, 32);
    l_run = l_run * alpha + s;
    m_run = m_new;
#pragma unroll
    for (int r = 0; r < 8; r++) { o0[r] *= alpha; o1[r] *= alpha; o2[r] *= alpha; o3[r] *= alpha; }

    // P^T (C-layout) -> 32x16 B-fragment: 2 shuffles per register
    v16bf pf;
#pragma unroll
    for (int r = 0; r < 8; r++) {
      float x0 = __shfl_xor(p0[r], 16, 32);
      float x1 = __shfl_xor(p1[r], 16, 32);
      pf[r]     = (__bf16)(lo ? p0[r] : x1);
      pf[r + 8] = (__bf16)(lo ? x0 : p1[r]);
    }

    // O^T += V^T(16dh x 32kv) * P^T(32kv x 16q)
    o0 = wmma_bf16(load_afrag(VTbase + (size_t)(qc) * SEQ,      kt, hl), pf, o0);
    o1 = wmma_bf16(load_afrag(VTbase + (size_t)(16 + qc) * SEQ, kt, hl), pf, o1);
    o2 = wmma_bf16(load_afrag(VTbase + (size_t)(32 + qc) * SEQ, kt, hl), pf, o2);
    o3 = wmma_bf16(load_afrag(VTbase + (size_t)(48 + qc) * SEQ, kt, hl), pf, o3);
  }

  float inv = 1.f / l_run;
  __bf16* orow = ATT + (size_t)(b * SEQ + qt * 16 + qc) * D_MODEL + h * HDIM + 8 * hl;
  union { uint4 q; __bf16 hh[8]; } pk;
#pragma unroll
  for (int r = 0; r < 8; r++) pk.hh[r] = (__bf16)(o0[r] * inv);
  *(uint4*)(orow + 0) = pk.q;
#pragma unroll
  for (int r = 0; r < 8; r++) pk.hh[r] = (__bf16)(o1[r] * inv);
  *(uint4*)(orow + 16) = pk.q;
#pragma unroll
  for (int r = 0; r < 8; r++) pk.hh[r] = (__bf16)(o2[r] * inv);
  *(uint4*)(orow + 32) = pk.q;
#pragma unroll
  for (int r = 0; r < 8; r++) pk.hh[r] = (__bf16)(o3[r] * inv);
  *(uint4*)(orow + 48) = pk.q;
}

// ------------------------- out-proj and gating -----------------------------

__global__ void __launch_bounds__(256)
k_outproj(const __bf16* __restrict__ ATT, const __bf16* __restrict__ WOUT,
          const float* __restrict__ outb,
          float* __restrict__ PROJ, __bf16* __restrict__ PROJB) {
  int gw   = (blockIdx.x * 256 + threadIdx.x) >> 5;
  int lane = threadIdx.x & 31, qc = lane & 15, hl = lane >> 4;
  int mt = gw >> 5, nt = gw & 31;
  v8f acc = gemm_tile_512(ATT + (size_t)(mt * 16 + qc) * D_MODEL,
                          WOUT + (size_t)(nt * 16 + qc) * D_MODEL, hl);
#pragma unroll
  for (int r = 0; r < 8; r++) {
    int row = mt * 16 + r + 8 * hl;
    int col = nt * 16 + qc;
    float v = acc[r] + outb[col];
    size_t idx = (size_t)row * D_MODEL + col;
    PROJ[idx]  = v;
    PROJB[idx] = (__bf16)v;
  }
}

__global__ void __launch_bounds__(256)
k_gate(const __bf16* __restrict__ PROJB, const __bf16* __restrict__ WGATE,
       const float* __restrict__ gateb, const float* __restrict__ PROJ,
       const float* __restrict__ residual, float* __restrict__ out) {
  int gw   = (blockIdx.x * 256 + threadIdx.x) >> 5;
  int lane = threadIdx.x & 31, qc = lane & 15, hl = lane >> 4;
  int mt = gw >> 5, nt = gw & 31;
  v8f acc = gemm_tile_512(PROJB + (size_t)(mt * 16 + qc) * D_MODEL,
                          WGATE + (size_t)(nt * 16 + qc) * D_MODEL, hl);
#pragma unroll
  for (int r = 0; r < 8; r++) {
    int row = mt * 16 + r + 8 * hl;
    int col = nt * 16 + qc;
    float gl = acc[r] + gateb[col];
    float g  = 1.f / (1.f + __expf(-gl));
    size_t idx = (size_t)row * D_MODEL + col;
    float a = PROJ[idx];
    out[idx] = g * a + (1.f - g) * residual[idx];
  }
}

// ------------------------------- launcher ----------------------------------

extern "C" void kernel_launch(void* const* d_in, const int* in_sizes, int n_in,
                              void* d_out, int out_size, void* d_ws, size_t ws_size,
                              hipStream_t stream) {
  (void)in_sizes; (void)n_in; (void)out_size; (void)ws_size;
  const float* dec   = (const float*)d_in[0];
  const float* enc   = (const float*)d_in[1];
  const float* qkvw  = (const float*)d_in[2];
  const float* outw  = (const float*)d_in[3];
  const float* outb  = (const float*)d_in[4];
  const float* gatew = (const float*)d_in[5];
  const float* gateb = (const float*)d_in[6];
  const float* lng   = (const float*)d_in[7];
  const float* lnb   = (const float*)d_in[8];
  float* out = (float*)d_out;

  char* w = (char*)d_ws;
  size_t off = 0;
  auto take = [&](size_t bytes) -> char* {
    char* p = w + off;
    off += (bytes + 255) & ~(size_t)255;
    return p;
  };
  __bf16* LNQ   = (__bf16*)take((size_t)ROWS * D_MODEL * 2);      // 4 MB
  __bf16* LNKV  = (__bf16*)take((size_t)ROWS * D_MODEL * 2);      // 4 MB
  __bf16* WQKV  = (__bf16*)take((size_t)3 * 512 * 512 * 2);       // 1.5 MB
  __bf16* WOUT  = (__bf16*)take((size_t)512 * 512 * 2);           // 0.5 MB
  __bf16* WGATE = (__bf16*)take((size_t)512 * 512 * 2);           // 0.5 MB
  __bf16* QB    = (__bf16*)take((size_t)ROWS * D_MODEL * 2);      // 4 MB
  __bf16* KB    = (__bf16*)take((size_t)ROWS * D_MODEL * 2);      // 4 MB
  __bf16* VTb   = (__bf16*)take((size_t)ROWS * D_MODEL * 2);      // 4 MB
  float*  BIAS  = (float*) take((size_t)SEQ * SEQ * 4);           // 16 MB (L2-resident)
  __bf16* ATT   = (__bf16*)take((size_t)ROWS * D_MODEL * 2);      // 4 MB
  float*  PROJ  = (float*) take((size_t)ROWS * D_MODEL * 4);      // 8 MB
  __bf16* PROJB = (__bf16*)take((size_t)ROWS * D_MODEL * 2);      // 4 MB

  k_cvt <<<3072, 256, 0, stream>>>(qkvw,  WQKV,  3 * 512 * 512);
  k_cvt <<<1024, 256, 0, stream>>>(outw,  WOUT,  512 * 512);
  k_cvt <<<1024, 256, 0, stream>>>(gatew, WGATE, 512 * 512);
  k_bias<<<16384, 256, 0, stream>>>(BIAS);
  k_ln  <<<1024, 256, 0, stream>>>(dec, enc, lng, lnb, LNQ, LNKV);
  k_qkv <<<3072, 256, 0, stream>>>(LNQ, LNKV, WQKV, QB, KB, VTb);
  k_attn<<<256,  256, 0, stream>>>(QB, KB, VTb, BIAS, ATT);
  k_outproj<<<1024, 256, 0, stream>>>(ATT, WOUT, outb, PROJ, PROJB);
  k_gate<<<1024, 256, 0, stream>>>(PROJB, WGATE, gateb, PROJ, dec, out);
}
